// EVLDecoder_2156073582807
// MI455X (gfx1250) — matrix operand order repaired and checked by reading, not compiled, not run on hardware
//
#include <hip/hip_runtime.h>
#include <hip/hip_bf16.h>
#include <math.h>
#include <stdint.h>

#define L_   4
#define C_   768
#define H_   12
#define T_   64
#define N_   1024
#define MLP_ 3072
#define EPS_ 1e-5f

typedef _Float16 h16;
typedef _Float16 v4h  __attribute__((ext_vector_type(4)));
typedef _Float16 v8h  __attribute__((ext_vector_type(8)));
typedef _Float16 v16h __attribute__((ext_vector_type(16)));
typedef float    v8f  __attribute__((ext_vector_type(8)));
typedef int      v4i  __attribute__((__vector_size__(4 * sizeof(int))));

// ---- CDNA5 async global->LDS copy (ASYNCcnt-tracked), with safe fallback ----
#if defined(__has_builtin)
#if __has_builtin(__builtin_amdgcn_global_load_async_to_lds_b128) && \
    __has_builtin(__builtin_amdgcn_s_wait_asynccnt)
#define HAS_ASYNC_LDS 1
#endif
#endif
#ifndef HAS_ASYNC_LDS
#define HAS_ASYNC_LDS 0
#endif

typedef __attribute__((address_space(1))) v4i as1_v4i;
typedef __attribute__((address_space(3))) v4i as3_v4i;

__device__ __forceinline__ void async_copy16(const h16* g, h16* l) {
#if HAS_ASYNC_LDS
  __builtin_amdgcn_global_load_async_to_lds_b128(
      (as1_v4i*)(uintptr_t)g, (as3_v4i*)(uint32_t)(uintptr_t)l, 0, 0);
#else
  *(v8h*)l = *(const v8h*)g;
#endif
}

// ---------------------------------------------------------------- elementwise
__global__ void k_convert_f16(const float* __restrict__ src, h16* __restrict__ dst, int n) {
  int i = blockIdx.x * blockDim.x + threadIdx.x;
  int stride = gridDim.x * blockDim.x;
  for (; i < n; i += stride) dst[i] = (h16)src[i];
}

// dst[n*K + k] = (f16) src[k*Nn + n]   (pre-transpose weights -> BT[Nn][K])
__global__ void k_convert_f16_t(const float* __restrict__ src, h16* __restrict__ dst,
                                int K, int Nn) {
  int i = blockIdx.x * blockDim.x + threadIdx.x;   // over K*Nn, write-contiguous
  if (i >= K * Nn) return;
  int n = i / K, k = i - n * K;
  dst[i] = (h16)src[(size_t)k * Nn + n];
}

__global__ void k_init_x(const float* __restrict__ cls, float* __restrict__ x) {
  int i = blockIdx.x * blockDim.x + threadIdx.x;   // N_*C_ threads
  x[i] = cls[i % C_];
}

__global__ void k_copy_out(const float* __restrict__ x, float* __restrict__ out, int n) {
  int i = blockIdx.x * blockDim.x + threadIdx.x;
  if (i < n) out[i] = x[i];
}

// ------------------------------------------------- conv + residual + pos + LN3
__global__ __launch_bounds__(256) void k_conv_ln3(
    const float* __restrict__ in,   // [N,T,C]
    const float* __restrict__ cw,   // [C,3]  (this layer)
    const float* __restrict__ cb,   // [C]
    const float* __restrict__ pos,  // [T,C]
    const float* __restrict__ g, const float* __restrict__ b,
    h16* __restrict__ yn)           // [N*T, C] f16
{
  __shared__ float red[256];
  __shared__ float red2[256];
  const int bt  = blockIdx.x;
  const int t   = bt % T_;
  const int tid = threadIdx.x;
  const float* row  = in + (size_t)bt * C_;
  const float* rowm = (t > 0)     ? row - C_ : nullptr;
  const float* rowp = (t < T_-1)  ? row + C_ : nullptr;

  float y[3]; float s = 0.f, s2 = 0.f;
  #pragma unroll
  for (int j = 0; j < 3; ++j) {
    int c = tid + j * 256;
    float xc = row[c];
    float xm = rowm ? rowm[c] : 0.f;
    float xp = rowp ? rowp[c] : 0.f;
    const float* w = cw + c * 3;
    float v = xc + xm * w[0] + xc * w[1] + xp * w[2] + cb[c] + pos[t * C_ + c];
    y[j] = v; s += v; s2 += v * v;
  }
  red[tid] = s; red2[tid] = s2;
  __syncthreads();
  for (int off = 128; off > 0; off >>= 1) {
    if (tid < off) { red[tid] += red[tid + off]; red2[tid] += red2[tid + off]; }
    __syncthreads();
  }
  float mu  = red[0]  * (1.f / C_);
  float var = red2[0] * (1.f / C_) - mu * mu;
  float rs  = rsqrtf(var + EPS_);
  #pragma unroll
  for (int j = 0; j < 3; ++j) {
    int c = tid + j * 256;
    yn[(size_t)bt * C_ + c] = (h16)((y[j] - mu) * rs * g[c] + b[c]);
  }
}

// ------------------------------------------------------------- LN over x rows
__global__ __launch_bounds__(256) void k_ln_rows(
    const float* __restrict__ x, const float* __restrict__ g, const float* __restrict__ b,
    h16* __restrict__ out)
{
  __shared__ float red[256];
  __shared__ float red2[256];
  const int n = blockIdx.x, tid = threadIdx.x;
  const float* row = x + (size_t)n * C_;
  float y[3]; float s = 0.f, s2 = 0.f;
  #pragma unroll
  for (int j = 0; j < 3; ++j) {
    int c = tid + j * 256;
    float v = row[c];
    y[j] = v; s += v; s2 += v * v;
  }
  red[tid] = s; red2[tid] = s2;
  __syncthreads();
  for (int off = 128; off > 0; off >>= 1) {
    if (tid < off) { red[tid] += red[tid + off]; red2[tid] += red2[tid + off]; }
    __syncthreads();
  }
  float mu  = red[0]  * (1.f / C_);
  float var = red2[0] * (1.f / C_) - mu * mu;
  float rs  = rsqrtf(var + EPS_);
  #pragma unroll
  for (int j = 0; j < 3; ++j) {
    int c = tid + j * 256;
    out[(size_t)n * C_ + c] = (h16)((y[j] - mu) * rs * g[c] + b[c]);
  }
}

// --------------------------------------------------------------- WMMA GEMM
// C[M,Nn] = A[M,K] (f16 row-major) * BT[Nn,K]^T (f16, pre-transposed) + bias[Nn]
// EPI 0: store f16       EPI 1: QuickGELU -> f16       EPI 2: f32 resid + out
#define LDS_TILE (128 * 40)

__device__ __forceinline__ void stage_tiles(
    const h16* __restrict__ A, const h16* __restrict__ BT,
    h16* ldsA, h16* ldsB, int bM, int bN, int kc, int K, int tid)
{
  const int r = tid >> 2;          // 0..63
  const int c = (tid & 3) * 8;     // 0,8,16,24
  async_copy16(A  + (size_t)(bM + r)      * K + kc + c, ldsA + r * 40 + c);
  async_copy16(A  + (size_t)(bM + r + 64) * K + kc + c, ldsA + (r + 64) * 40 + c);
  async_copy16(BT + (size_t)(bN + r)      * K + kc + c, ldsB + r * 40 + c);
  async_copy16(BT + (size_t)(bN + r + 64) * K + kc + c, ldsB + (r + 64) * 40 + c);
}

template<int EPI>
__global__ __launch_bounds__(256) void wmma_gemm(
    const h16* __restrict__ A, const h16* __restrict__ BT,
    const float* __restrict__ bias,
    h16* __restrict__ outH, float* __restrict__ outF, const float* __restrict__ resid,
    int M, int Nn, int K)
{
  __shared__ __align__(16) h16 ldsA[2][LDS_TILE];   // [row][k], stride 40
  __shared__ __align__(16) h16 ldsB[2][LDS_TILE];   // [col][k], stride 40
  const int tid  = threadIdx.x;
  const int lane = tid & 31;
  const int wave = tid >> 5;
  const int wm = wave & 3, wn = wave >> 2;          // 4 x 2 wave grid
  const int bM = blockIdx.y * 128;
  const int bN = blockIdx.x * 128;

  v8f acc[2][4];
  #pragma unroll
  for (int i = 0; i < 2; ++i)
    #pragma unroll
    for (int j = 0; j < 4; ++j)
      #pragma unroll
      for (int r = 0; r < 8; ++r) acc[i][j][r] = 0.f;

  // prologue: stage first K-chunk into buffer 0
  stage_tiles(A, BT, ldsA[0], ldsB[0], bM, bN, 0, K, tid);

  int buf = 0;
  for (int kc = 0; kc < K; kc += 32, buf ^= 1) {
    const bool has_next = (kc + 32 < K);
    if (has_next)
      stage_tiles(A, BT, ldsA[buf ^ 1], ldsB[buf ^ 1], bM, bN, kc + 32, K, tid);
#if HAS_ASYNC_LDS
    if (has_next) __builtin_amdgcn_s_wait_asynccnt(4);  // current buffer landed
    else          __builtin_amdgcn_s_wait_asynccnt(0);
#endif
    __syncthreads();

    // ---- fragments (ISA 7.12.2 layouts)
    v16h aF[2], bF[4];
    #pragma unroll
    for (int tm = 0; tm < 2; ++tm) {
      int row = wm * 32 + tm * 16 + (lane & 15);
      int kb  = (lane >> 4) * 8;               // lanes 0-15: K 0..7/16..23; 16-31: 8..15/24..31
      v8h lo = *(const v8h*)(&ldsA[buf][row * 40 + kb]);
      v8h hi = *(const v8h*)(&ldsA[buf][row * 40 + kb + 16]);
      aF[tm] = __builtin_shufflevector(lo, hi, 0,1,2,3,4,5,6,7,8,9,10,11,12,13,14,15);
    }
    #pragma unroll
    for (int tn = 0; tn < 4; ++tn) {
      int col = wn * 64 + tn * 16 + (lane & 15);
      int kb  = (lane >> 4) * 8;               // lanes 0-15: K 0..7/16..23; 16-31: 8..15/24..31
      v8h lo = *(const v8h*)(&ldsB[buf][col * 40 + kb]);
      v8h hi = *(const v8h*)(&ldsB[buf][col * 40 + kb + 16]);
      bF[tn] = __builtin_shufflevector(lo, hi, 0,1,2,3,4,5,6,7,8,9,10,11,12,13,14,15);
    }
    #pragma unroll
    for (int tm = 0; tm < 2; ++tm)
      #pragma unroll
      for (int tn = 0; tn < 4; ++tn)
        acc[tm][tn] = __builtin_amdgcn_wmma_f32_16x16x32_f16(
            false, aF[tm], false, bF[tn], (short)0, acc[tm][tn], false, false);
    __syncthreads();
  }

  // ---- epilogue
  #pragma unroll
  for (int tm = 0; tm < 2; ++tm)
    #pragma unroll
    for (int tn = 0; tn < 4; ++tn) {
      int col   = bN + wn * 64 + tn * 16 + (lane & 15);
      int rbase = bM + wm * 32 + tm * 16 + (lane >> 4) * 8;
      float bv = bias[col];
      #pragma unroll
      for (int r = 0; r < 8; ++r) {
        size_t idx = (size_t)(rbase + r) * Nn + col;
        float v = acc[tm][tn][r] + bv;
        if (EPI == 1) v = v / (1.f + __expf(-1.702f * v));   // QuickGELU
        if (EPI == 2) outF[idx] = resid[idx] + v;
        else          outH[idx] = (h16)v;
      }
    }
}

// ---------------------------------------------------- masked cross-attention
__global__ __launch_bounds__(256) void k_attn(
    const h16* __restrict__ q,    // [N,C] f16
    const h16* __restrict__ k,    // [N,T,C] f16
    const h16* __restrict__ v,    // [N,T,C] f16
    const int* __restrict__ mask, // [N,T]
    h16* __restrict__ mix)        // [N,C] f16
{
  __shared__ __align__(16) h16 qs[C_];
  __shared__ float affs[T_ * H_];
  __shared__ float mh[H_], sh[H_];
  const int n = blockIdx.x, tid = threadIdx.x;
  if (tid < C_ / 8)   // 96 lanes stage q as b128
    ((v8h*)qs)[tid] = ((const v8h*)(q + (size_t)n * C_))[tid];
  __syncthreads();

  const float scale = 0.125f;  // (C/H)^-0.5 = 64^-0.5
  for (int p = tid; p < T_ * H_; p += 256) {
    int t = p / H_, h = p % H_;
    float a;
    if (mask[n * T_ + t]) {
      const v8h* kp = (const v8h*)(k + ((size_t)n * T_ + t) * C_ + h * 64);
      const v8h* qp = (const v8h*)(qs + h * 64);
      float s = 0.f;
      #pragma unroll
      for (int d8 = 0; d8 < 8; ++d8) {       // 8 x b128 loads
        v8h kv = kp[d8], qv = qp[d8];
        #pragma unroll
        for (int j = 0; j < 8; ++j) s += (float)qv[j] * (float)kv[j];
      }
      a = s * scale;
    } else {
      a = -__builtin_inff();
    }
    affs[p] = a;
  }
  __syncthreads();
  if (tid < H_) {
    float m = -__builtin_inff();
    for (int t = 0; t < T_; ++t) m = fmaxf(m, affs[t * H_ + tid]);
    float s = 0.f;
    for (int t = 0; t < T_; ++t) s += __expf(affs[t * H_ + tid] - m);
    mh[tid] = m; sh[tid] = 1.f / s;
  }
  __syncthreads();
  for (int p = tid; p < T_ * H_; p += 256) {
    int h = p % H_;
    affs[p] = __expf(affs[p] - mh[h]) * sh[h];
  }
  __syncthreads();
  // mix: each of 192 lanes owns 4 contiguous channels -> b64 loads per timestep
  if (tid < C_ / 4) {
    const int p = tid * 4;
    const int h = p >> 6;
    const h16* vp = v + (size_t)n * T_ * C_ + p;
    float a0 = 0.f, a1 = 0.f, a2 = 0.f, a3 = 0.f;
    #pragma unroll 8
    for (int t = 0; t < T_; ++t) {
      v4h vv = *(const v4h*)(vp + (size_t)t * C_);
      float a = affs[t * H_ + h];
      a0 += a * (float)vv[0];
      a1 += a * (float)vv[1];
      a2 += a * (float)vv[2];
      a3 += a * (float)vv[3];
    }
    v4h o; o[0] = (h16)a0; o[1] = (h16)a1; o[2] = (h16)a2; o[3] = (h16)a3;
    *(v4h*)(mix + (size_t)n * C_ + p) = o;
  }
}

// ------------------------------------------------------------------- driver
extern "C" void kernel_launch(void* const* d_in, const int* in_sizes, int n_in,
                              void* d_out, int out_size, void* d_ws, size_t ws_size,
                              hipStream_t stream)
{
  const float* in_features = (const float*)d_in[0];
  const int*   vmask       = (const int*)d_in[1];
  const float* cls         = (const float*)d_in[2];
  const float* conv_w      = (const float*)d_in[3];
  const float* conv_b      = (const float*)d_in[4];
  const float* pos         = (const float*)d_in[5];
  const float* wq = (const float*)d_in[6];  const float* bq = (const float*)d_in[7];
  const float* wk = (const float*)d_in[8];  const float* bk = (const float*)d_in[9];
  const float* wv = (const float*)d_in[10]; const float* bv = (const float*)d_in[11];
  const float* wo = (const float*)d_in[12]; const float* bo = (const float*)d_in[13];
  const float* f1w = (const float*)d_in[14]; const float* f1b = (const float*)d_in[15];
  const float* f2w = (const float*)d_in[16]; const float* f2b = (const float*)d_in[17];
  const float* l1g = (const float*)d_in[18]; const float* l1b = (const float*)d_in[19];
  const float* l2g = (const float*)d_in[20]; const float* l2b = (const float*)d_in[21];
  const float* l3g = (const float*)d_in[22]; const float* l3b = (const float*)d_in[23];

  char* ws = (char*)d_ws;
  size_t off = 0;
  auto alloc = [&](size_t bytes) -> char* {
    char* p = ws + off;
    off = (off + bytes + 255) & ~(size_t)255;
    return p;
  };

  float* x   = (float*)alloc((size_t)N_ * C_ * 4);
  h16* xn    = (h16*)alloc((size_t)N_ * C_ * 2);
  h16* qbuf  = (h16*)alloc((size_t)N_ * C_ * 2);
  h16* mix   = (h16*)alloc((size_t)N_ * C_ * 2);
  h16* hbuf  = (h16*)alloc((size_t)N_ * MLP_ * 2);
  h16* yn    = (h16*)alloc((size_t)N_ * T_ * C_ * 2);
  h16* kbuf  = (h16*)alloc((size_t)N_ * T_ * C_ * 2);
  h16* vbuf  = (h16*)alloc((size_t)N_ * T_ * C_ * 2);
  h16* wq_h  = (h16*)alloc((size_t)L_ * C_ * C_ * 2);   // transposed [out][in]
  h16* wk_h  = (h16*)alloc((size_t)L_ * C_ * C_ * 2);
  h16* wv_h  = (h16*)alloc((size_t)L_ * C_ * C_ * 2);
  h16* wo_h  = (h16*)alloc((size_t)L_ * C_ * C_ * 2);
  h16* f1_h  = (h16*)alloc((size_t)L_ * C_ * MLP_ * 2); // transposed [MLP][C]
  h16* f2_h  = (h16*)alloc((size_t)L_ * MLP_ * C_ * 2); // transposed [C][MLP]

  // f32 -> f16 with transpose: dst layer layout is BT[Nn][K]
  auto cvtT = [&](const float* s, h16* d, int K, int Nn) {
    int total = K * Nn;
    int blocks = (total + 255) / 256;
    for (int l = 0; l < L_; ++l)
      k_convert_f16_t<<<blocks, 256, 0, stream>>>(
          s + (size_t)l * total, d + (size_t)l * total, K, Nn);
  };
  cvtT(wq,  wq_h, C_, C_);
  cvtT(wk,  wk_h, C_, C_);
  cvtT(wv,  wv_h, C_, C_);
  cvtT(wo,  wo_h, C_, C_);
  cvtT(f1w, f1_h, C_, MLP_);
  cvtT(f2w, f2_h, MLP_, C_);

  k_init_x<<<(N_ * C_) / 256, 256, 0, stream>>>(cls, x);

  for (int l = 0; l < L_; ++l) {
    // y = in + dwconv(in) + pos ; yn = LN3(y)  (f16)
    k_conv_ln3<<<N_ * T_, 256, 0, stream>>>(
        in_features, conv_w + (size_t)l * C_ * 3, conv_b + (size_t)l * C_,
        pos + (size_t)l * T_ * C_, l3g + (size_t)l * C_, l3b + (size_t)l * C_, yn);

    // K, V projections: [65536,768] @ [768,768]
    dim3 gKV(C_ / 128, (N_ * T_) / 128);
    wmma_gemm<0><<<gKV, 256, 0, stream>>>(yn, wk_h + (size_t)l * C_ * C_,
        bk + (size_t)l * C_, kbuf, nullptr, nullptr, N_ * T_, C_, C_);
    wmma_gemm<0><<<gKV, 256, 0, stream>>>(yn, wv_h + (size_t)l * C_ * C_,
        bv + (size_t)l * C_, vbuf, nullptr, nullptr, N_ * T_, C_, C_);

    // Q projection on LN1(x)
    k_ln_rows<<<N_, 256, 0, stream>>>(x, l1g + (size_t)l * C_, l1b + (size_t)l * C_, xn);
    dim3 gQ(C_ / 128, N_ / 128);
    wmma_gemm<0><<<gQ, 256, 0, stream>>>(xn, wq_h + (size_t)l * C_ * C_,
        bq + (size_t)l * C_, qbuf, nullptr, nullptr, N_, C_, C_);

    // masked softmax attention (Lq = 1)
    k_attn<<<N_, 256, 0, stream>>>(qbuf, kbuf, vbuf, vmask, mix);

    // x += mix @ wo + bo   (f32 residual, in place)
    wmma_gemm<2><<<gQ, 256, 0, stream>>>(mix, wo_h + (size_t)l * C_ * C_,
        bo + (size_t)l * C_, nullptr, x, x, N_, C_, C_);

    // MLP: h = QuickGELU(LN2(x) @ fc1 + b1); x += h @ fc2 + b2
    k_ln_rows<<<N_, 256, 0, stream>>>(x, l2g + (size_t)l * C_, l2b + (size_t)l * C_, xn);
    dim3 gF1(MLP_ / 128, N_ / 128);
    wmma_gemm<1><<<gF1, 256, 0, stream>>>(xn, f1_h + (size_t)l * C_ * MLP_,
        f1b + (size_t)l * MLP_, hbuf, nullptr, nullptr, N_, MLP_, C_);
    dim3 gF2(C_ / 128, N_ / 128);
    wmma_gemm<2><<<gF2, 256, 0, stream>>>(hbuf, f2_h + (size_t)l * MLP_ * C_,
        f2b + (size_t)l * C_, nullptr, x, x, N_, C_, MLP_);
  }

  k_copy_out<<<(out_size + 255) / 256, 256, 0, stream>>>(x, (float*)d_out, out_size);
}
